// Communication_13932873908844
// MI455X (gfx1250) — matrix operand order repaired and checked by reading, not compiled, not run on hardware
//
#include <hip/hip_runtime.h>

// ---------------------------------------------------------------------------
// Communication mask kernel for MI455X (gfx1250)
//   conf: (8,6,4,512,512) f32   warp: (8,1,512,512) f32   gk: (1,1,5,5) f32
//   out:  mask (48*512*512 f32) ++ rate (1 f32)
// Pipeline: [sigmoid∘max ∘ warp ∘ separable 5x5 gauss (WMMA) + radix hist]
//           -> per-row threshold select -> [async-LDS staged mask emit]
// ---------------------------------------------------------------------------

#define HDIM 512
#define WDIM 512
#define CDIM 4
#define LDIM 6
#define NROW 48                    // B*L
#define HW   (HDIM * WDIM)        // 262144 = 2^18
#define KSEL (HW / 2)             // top-k K = 131072 (=> rate == 0.5 exactly)
#define NBIN 4096

// conv-kernel tiling: 8 waves/block, each wave one 16x16 WMMA tile,
// block region 64x32, shared halo 68x36 (stride padded to 69)
#define RGNW 64
#define RGNH 32
#define HALW 68
#define HALH 36
#define HSTR 69

typedef __attribute__((ext_vector_type(16))) _Float16 v16h;
typedef __attribute__((ext_vector_type(8)))  float    v8f;

// ---------------------------------------------------------------------------
// Kernel 1: fused sigmoid(max_c)*warp + separable Gaussian via WMMA,
//           plus per-block radix histogram of the produced values.
// ---------------------------------------------------------------------------
__global__ __launch_bounds__(256) void fused_smooth_hist_kernel(
    const float* __restrict__ conf, const float* __restrict__ warp,
    const float* __restrict__ gk, float* __restrict__ sm,
    unsigned* __restrict__ ghist)
{
    __shared__ float    mlds[HALH * HSTR];   // 36x68 halo region (padded stride)
    __shared__ unsigned hist[NBIN];

    const int tid  = threadIdx.x;            // 0..255
    const int lane = tid & 31;               // wave32 lane
    const int wv   = tid >> 5;               // 0..7: wave -> 16x16 tile
    const int z    = blockIdx.z;             // b*L + l  (one (b,l) row per block)
    const int b    = z / LDIM;
    const long long confBase = (long long)z * CDIM * HW;
    const long long warpBase = (long long)b * HW;
    const int x0 = blockIdx.x * RGNW;        // block region origin (image coords)
    const int y0 = blockIdx.y * RGNH;

    // Separable 1-D factors recovered from the (separable) input 5x5 kernel.
    float wy[5], wx[5];
    const float cker = gk[12];
#pragma unroll
    for (int t = 0; t < 5; ++t) {
        wy[t] = gk[t * 5 + 2] / cker;        // peak-normalized vertical taps
        wx[t] = gk[10 + t];                  // horizontal taps (carry 1/(2*pi*s))
    }

    for (int i = tid; i < NBIN; i += 256) hist[i] = 0u;

    // Stage 1: m = sigmoid(max over C) * warp for the 36x68 halo (zero pad).
    for (int i = tid; i < HALH * HALW; i += 256) {
        const int r = i / HALW, c = i % HALW;
        const int y = y0 - 2 + r, x = x0 - 2 + c;
        float m = 0.0f;
        if ((unsigned)y < HDIM && (unsigned)x < WDIM) {
            const long long p = (long long)y * WDIM + x;
            float v0 = conf[confBase + 0 * HW + p];
            float v1 = conf[confBase + 1 * HW + p];
            float v2 = conf[confBase + 2 * HW + p];
            float v3 = conf[confBase + 3 * HW + p];
            float mx = fmaxf(fmaxf(v0, v1), fmaxf(v2, v3));
            m = (1.0f / (1.0f + expf(-mx))) * warp[warpBase + p];  // sigmoid∘max
        }
        mlds[r * HSTR + c] = m;
    }
    __syncthreads();

    // Stage 2: vertical 5-tap -> A fragment (16-bit A 16x32 layout, K=20 used).
    const int hi  = lane >> 4;
    const int row = lane & 15;               // M for A, N for B/D
    const int tox = (wv & 3) * 16;           // tile origin within block region
    const int toy = (wv >> 2) * 16;
    v16h a;
#pragma unroll
    for (int e = 0; e < 16; ++e) {
        const int k = (e & 7) + ((e >> 3) << 4) + 8 * hi;   // ISA A layout K-index
        float acc = 0.0f;
        if (k < 20) {
#pragma unroll
            for (int t = 0; t < 5; ++t)
                acc += wy[t] * mlds[(toy + row + t) * HSTR + (tox + k)];
        }
        a[e] = (_Float16)acc;
    }

    // Banded horizontal matrix B[k][n] = wx[k-n] (16-bit B 32x16 layout).
    v16h bb;
#pragma unroll
    for (int e = 0; e < 16; ++e) {
        const int k = (e & 7) + ((e >> 3) << 4) + 8 * hi;
        const int d = k - row;
        bb[e] = (_Float16)((d >= 0 && d <= 4) ? wx[d] : 0.0f);
    }

    v8f c = {};
    // D = A x B : one tensor op does the whole horizontal conv for the tile.
    c = __builtin_amdgcn_wmma_f32_16x16x32_f16(
            /*neg_a=*/false, a, /*neg_b=*/false, bb,
            /*c_mod=*/(short)0, c, /*reuse_a=*/false, /*reuse_b=*/false);

    // Stage 3: store smoothed values + bin them (positive floats: bit order).
    float* dst = sm + (long long)z * HW;
#pragma unroll
    for (int r = 0; r < 8; ++r) {
        const int M = r + 8 * hi;                          // D layout row
        const float v = c[r];
        dst[(long long)(y0 + toy + M) * WDIM + (x0 + tox + row)] = v;
        atomicAdd(&hist[(__float_as_uint(v) >> 19) & (NBIN - 1)], 1u);
    }
    __syncthreads();

    // Flush block-local histogram (all outputs of this block share row z).
    for (int i = tid; i < NBIN; i += 256) {
        const unsigned h = hist[i];
        if (h) atomicAdd(&ghist[z * NBIN + i], h);
    }
}

// ---------------------------------------------------------------------------
// Kernel 2: per-row suffix scan -> threshold bin + #needed from the tie bin.
// Also emits the (exactly constant) communication rate K/(H*W) = 0.5.
// ---------------------------------------------------------------------------
__global__ void select_kernel(const unsigned* __restrict__ ghist,
                              int* __restrict__ tbin,
                              unsigned* __restrict__ need,
                              unsigned* __restrict__ ctr,
                              float* __restrict__ out, long long total)
{
    const int r = blockIdx.x * blockDim.x + threadIdx.x;
    if (r == 0) out[total] = 0.5f;           // rate (computed pre-ego in ref)
    if (r >= NROW) return;
    const unsigned* h = ghist + r * NBIN;
    unsigned cum = 0; int t = 0; unsigned nd = 0;
    for (int bn = NBIN - 1; bn >= 0; --bn) {
        const unsigned hb = h[bn];
        if (cum + hb >= (unsigned)KSEL) { t = bn; nd = (unsigned)KSEL - cum; break; }
        cum += hb;
    }
    tbin[r] = t; need[r] = nd; ctr[r] = 0u;
}

// ---------------------------------------------------------------------------
// Kernel 3: emit mask. Non-ego rows stage sm chunks Global->LDS with the
// CDNA5 async-copy path (ASYNCcnt), then threshold; ego rows write 1s only.
// ---------------------------------------------------------------------------
__global__ __launch_bounds__(256) void mask_kernel(
    const float* __restrict__ sm, const int* __restrict__ tbin,
    const unsigned* __restrict__ need, unsigned* __restrict__ ctr,
    float* __restrict__ out)
{
    __shared__ unsigned sdata[4096];         // 16 KB staged chunk (LDS offset 0)

    const int tid  = threadIdx.x;
    const int chnk = blockIdx.x;             // 0..63 chunks of 4096 elems
    const int rrow = blockIdx.y;             // 0..47
    const long long base = (long long)rrow * HW + chnk * 4096;
    float* dst = out + base;

    if ((rrow % LDIM) == 0) {                // ego agent keeps everything
#pragma unroll
        for (int i = 0; i < 16; ++i) dst[i * 256 + tid] = 1.0f;
        return;
    }

    const float* src = sm + base;
#pragma unroll
    for (int i = 0; i < 16; ++i) {
        const int e = i * 256 + tid;
        const unsigned ldsoff = (unsigned)e * 4u;
        const float* g = src + e;
        asm volatile("global_load_async_to_lds_b32 %0, %1, off"
                     :: "v"(ldsoff), "v"(g) : "memory");
    }
    asm volatile("s_wait_asynccnt 0x0" ::: "memory");
    __syncthreads();

    const int t = tbin[rrow];
    const unsigned nd = need[rrow];
#pragma unroll
    for (int i = 0; i < 16; ++i) {
        const int e = i * 256 + tid;
        const int bin = (int)(sdata[e] >> 19);
        float o;
        if (bin > t)       o = 1.0f;
        else if (bin == t) o = (atomicAdd(&ctr[rrow], 1u) < nd) ? 1.0f : 0.0f;
        else               o = 0.0f;
        dst[e] = o;
    }
}

__global__ void zero_kernel(unsigned* __restrict__ p, int n)
{
    const int i = blockIdx.x * 256 + threadIdx.x;
    if (i < n) p[i] = 0u;
}

// ---------------------------------------------------------------------------
extern "C" void kernel_launch(void* const* d_in, const int* in_sizes, int n_in,
                              void* d_out, int out_size, void* d_ws, size_t ws_size,
                              hipStream_t stream)
{
    const float* conf = (const float*)d_in[0];
    const float* warp = (const float*)d_in[1];
    const float* gk   = (const float*)d_in[2];
    float* out = (float*)d_out;

    // Workspace carve-up
    char* ws = (char*)d_ws;
    float*    sm    = (float*)ws;                                   // 48*HW f32
    unsigned* ghist = (unsigned*)(ws + (long long)NROW * HW * 4);   // 48*4096 u32
    int*      tbin  = (int*)((char*)ghist + NROW * NBIN * 4);
    unsigned* need  = (unsigned*)(tbin + NROW);
    unsigned* ctr   = need + NROW;

    const int nh = NROW * NBIN;
    hipLaunchKernelGGL(zero_kernel, dim3((nh + 255) / 256), dim3(256), 0, stream,
                       ghist, nh);

    // Fused conv (WMMA) + histogram: 8 waves/block over a 64x32 region.
    dim3 g1(WDIM / RGNW, HDIM / RGNH, NROW);
    hipLaunchKernelGGL(fused_smooth_hist_kernel, g1, dim3(256), 0, stream,
                       conf, warp, gk, sm, ghist);

    const long long total = (long long)NROW * HW;
    hipLaunchKernelGGL(select_kernel, dim3(1), dim3(64), 0, stream,
                       ghist, tbin, need, ctr, out, total);

    dim3 g3(HW / 4096, NROW);
    hipLaunchKernelGGL(mask_kernel, g3, dim3(256), 0, stream,
                       sm, tbin, need, ctr, out);
}
